// BondAwareMessagePassing_51943334478378
// MI455X (gfx1250) — matrix operand order repaired and checked by reading, not compiled, not run on hardware
//
#include <hip/hip_runtime.h>
#include <hip/hip_bf16.h>

// ---------- types ----------
typedef __bf16 bf16_t;
typedef bf16_t v16bf __attribute__((ext_vector_type(16)));
typedef bf16_t v8bf  __attribute__((ext_vector_type(8)));
typedef bf16_t v4bf  __attribute__((ext_vector_type(4)));
typedef float  v8f   __attribute__((ext_vector_type(8)));
typedef float  v4f   __attribute__((ext_vector_type(4)));

#define N_NODES 100000
#define E_EDGES 1600000
#define IN_D    64
#define BOND_D  16
#define HID     128
#define OUT_D   64
#define COMB    145
#define KPAD    160      // 145 padded to 5*32 for K-loop
#define NH      384      // 3*HID fused first layer
#define NW2     96       // 64 (x) + 16 (e) + 1 (p) padded to 6*16
#define WAVES   8
#define BLOCK   256
#define MT      2        // edge tiles (16 edges each) per wave iteration

// combine two contiguous 8x bf16 LDS chunks into one 16-half WMMA fragment
__device__ __forceinline__ v16bf lds_frag(const bf16_t* p0, const bf16_t* p1) {
    v8bf a = *(const v8bf*)p0;
    v8bf b = *(const v8bf*)p1;
    return __builtin_shufflevector(a, b, 0,1,2,3,4,5,6,7,8,9,10,11,12,13,14,15);
}

// SiLU with hardware fast reciprocal (v_rcp_f32) instead of IEEE division
__device__ __forceinline__ float silu_fast(float h) {
    return h * __builtin_amdgcn_rcpf(1.0f + __expf(-h));
}

__global__ __launch_bounds__(BLOCK, 1)
void egnn_zero_kernel(float4* __restrict__ out, int n4) {
    int i = blockIdx.x * blockDim.x + threadIdx.x;
    int stride = gridDim.x * blockDim.x;
    float4 z = {0.f, 0.f, 0.f, 0.f};
    for (; i < n4; i += stride) out[i] = z;
}

__global__ __launch_bounds__(BLOCK, 1)
void egnn_edge_kernel(const float* __restrict__ x,
                      const float* __restrict__ pos,
                      const float* __restrict__ edge_attr,
                      const int*   __restrict__ edge_index,
                      const float* __restrict__ W1x, const float* __restrict__ b1x,
                      const float* __restrict__ W2x, const float* __restrict__ b2x,
                      const float* __restrict__ W1p, const float* __restrict__ b1p,
                      const float* __restrict__ W2p, const float* __restrict__ b2p,
                      const float* __restrict__ W1e, const float* __restrict__ b1e,
                      const float* __restrict__ W2e, const float* __restrict__ b2e,
                      float* __restrict__ aggx,
                      float* __restrict__ aggp,
                      float* __restrict__ edgeu)
{
    // ---------------- LDS (~294 KB < 320 KB/WGP) ----------------
    __shared__ bf16_t sW1t[NH][KPAD];          // 122880 B  W1 fused, transposed
    __shared__ bf16_t sW2t[NW2][HID];          //  24576 B  W2 packed, transposed
    __shared__ float  sB1[NH];                 //   1536 B
    __shared__ float  sB2[NW2];                //    384 B
    __shared__ bf16_t sA[WAVES][MT][16][KPAD]; //  81920 B  per-wave A tiles
    __shared__ bf16_t sH[WAVES][MT][16][HID];  //  65536 B  per-wave hidden tiles
    __shared__ float  sRel[WAVES][MT][16][3];  //   3072 B
    __shared__ int    sCol[WAVES][MT][16];     //   1024 B

    const int tid = threadIdx.x;

    // -------- stage weights (fp32 -> bf16, transposed) --------
    for (int idx = tid; idx < NH * KPAD; idx += BLOCK) {
        int k = idx / NH, n = idx % NH;         // consecutive tid -> consecutive n (coalesced)
        float w = 0.0f;
        if (k < COMB) {
            if      (n < HID)     w = W1x[k * HID + n];
            else if (n < 2*HID)   w = W1p[k * HID + (n - HID)];
            else                  w = W1e[k * HID + (n - 2*HID)];
        }
        sW1t[n][k] = (bf16_t)w;
    }
    for (int idx = tid; idx < NW2 * HID; idx += BLOCK) {
        int k = idx / NW2, n = idx % NW2;
        float w = 0.0f;
        if      (n < OUT_D)           w = W2x[k * OUT_D + n];
        else if (n < OUT_D + BOND_D)  w = W2e[k * BOND_D + (n - OUT_D)];
        else if (n == 80)             w = W2p[k];
        sW2t[n][k] = (bf16_t)w;
    }
    for (int n = tid; n < NH; n += BLOCK)
        sB1[n] = (n < HID) ? b1x[n] : (n < 2*HID ? b1p[n - HID] : b1e[n - 2*HID]);
    for (int n = tid; n < NW2; n += BLOCK)
        sB2[n] = (n < OUT_D) ? b2x[n]
               : (n < 80 ? b2e[n - OUT_D] : (n == 80 ? b2p[0] : 0.0f));
    __syncthreads();

    const int lane = tid & 31;
    const int wv   = tid >> 5;
    const int m    = lane & 15;   // row within 16-edge tile / column within N-tile
    const int hi   = lane >> 4;   // half-wave selector

    bf16_t (*Aw)[16][KPAD] = sA[wv];
    bf16_t (*Hw)[16][HID]  = sH[wv];
    float  (*Relw)[16][3]  = sRel[wv];
    int    (*Colw)[16]     = sCol[wv];

    const int ntiles   = (E_EDGES + 15) / 16;
    const int npairs   = (ntiles + MT - 1) / MT;
    const int wvGlobal = blockIdx.x * WAVES + wv;
    const int wvStride = gridDim.x * WAVES;

    for (int pair = wvGlobal; pair < npairs; pair += wvStride) {
        const int ebase0 = pair * (16 * MT);

        // ---- build MT A tiles [16 x 160] bf16: [x[row] | x[col] | edge_attr | d2 | 0pad] ----
        #pragma unroll
        for (int mt = 0; mt < MT; ++mt) {
            const int ebase = ebase0 + mt * 16;
            int e = ebase + m; if (e >= E_EDGES) e = E_EDGES - 1;
            const int ri = edge_index[e];
            const int ci = edge_index[E_EDGES + e];
            {
                const v4f* src = (const v4f*)(x + (size_t)ri * IN_D + hi * 32);
                v4bf* dst = (v4bf*)&Aw[mt][m][hi * 32];
                #pragma unroll
                for (int q = 0; q < 8; ++q)
                    dst[q] = __builtin_convertvector(src[q], v4bf);  // packed cvt + b64 store
            }
            {
                const v4f* src = (const v4f*)(x + (size_t)ci * IN_D + hi * 32);
                v4bf* dst = (v4bf*)&Aw[mt][m][IN_D + hi * 32];
                #pragma unroll
                for (int q = 0; q < 8; ++q)
                    dst[q] = __builtin_convertvector(src[q], v4bf);
            }
            {
                const v4f* src = (const v4f*)(edge_attr + (size_t)e * BOND_D + hi * 8);
                v4bf* dst = (v4bf*)&Aw[mt][m][2*IN_D + hi * 8];
                #pragma unroll
                for (int q = 0; q < 2; ++q)
                    dst[q] = __builtin_convertvector(src[q], v4bf);
            }
            if (hi) {  // lanes 16..31: rel_pos / dist^2 / col index for all 16 edges
                float prx = pos[(size_t)ri*3+0], pry = pos[(size_t)ri*3+1], prz = pos[(size_t)ri*3+2];
                float pcx = pos[(size_t)ci*3+0], pcy = pos[(size_t)ci*3+1], pcz = pos[(size_t)ci*3+2];
                float rx = prx - pcx, ry = pry - pcy, rz = prz - pcz;
                float d2 = rx*rx + ry*ry + rz*rz;
                Relw[mt][m][0] = rx; Relw[mt][m][1] = ry; Relw[mt][m][2] = rz;
                Colw[mt][m] = ci;
                v4bf pad0 = { (bf16_t)d2, (bf16_t)0.f, (bf16_t)0.f, (bf16_t)0.f };
                v4bf z4   = { (bf16_t)0.f, (bf16_t)0.f, (bf16_t)0.f, (bf16_t)0.f };
                v4bf* dst = (v4bf*)&Aw[mt][m][144];
                dst[0] = pad0; dst[1] = z4; dst[2] = z4; dst[3] = z4;   // d2 + zero pad to 160
            }
        }
        // intra-wave LDS visibility (cross-lane A/Rel/Col reads follow)
        asm volatile("s_wait_dscnt 0" ::: "memory");

        // ---- A fragments (reused across all 3 phases, 8 N-tiles each) ----
        v16bf aF[MT][5];
        #pragma unroll
        for (int mt = 0; mt < MT; ++mt)
            #pragma unroll
            for (int kc = 0; kc < 5; ++kc) {
                int k0 = kc * 32 + hi * 8;
                aF[mt][kc] = lds_frag(&Aw[mt][m][k0], &Aw[mt][m][k0 + 16]);
            }

        #pragma unroll
        for (int phase = 0; phase < 3; ++phase) {   // 0:x  1:p  2:e
            const int nbase = phase * HID;
            // ------- GEMM1: [16x160] @ [160x128], B fragment shared across MT tiles -------
            #pragma unroll
            for (int j = 0; j < 8; ++j) {
                const int ng = nbase + j * 16 + m;
                v8f c[MT] = {};                      // zero init -> inline-0 SRC2 on first WMMA
                const bf16_t* wrow = sW1t[ng];
                #pragma unroll
                for (int kc = 0; kc < 5; ++kc) {
                    int k0 = kc * 32 + hi * 8;
                    v16bf bF = lds_frag(&wrow[k0], &wrow[k0 + 16]);
                    #pragma unroll
                    for (int mt = 0; mt < MT; ++mt)
                        c[mt] = __builtin_amdgcn_wmma_f32_16x16x32_bf16(
                                    false, aF[mt][kc], false, bF, (short)0, c[mt], false, false);
                }
                const float bias = sB1[ng];          // bias folded into SiLU epilogue
                const int colh = j * 16 + m;
                const int mrow = hi * 8;
                #pragma unroll
                for (int mt = 0; mt < MT; ++mt)
                    #pragma unroll
                    for (int r = 0; r < 8; ++r)
                        Hw[mt][mrow + r][colh] = (bf16_t)silu_fast(c[mt][r] + bias);
            }
            asm volatile("s_wait_dscnt 0" ::: "memory");

            // ------- GEMM2 A fragments from hidden tiles -------
            v16bf a2[MT][4];
            #pragma unroll
            for (int mt = 0; mt < MT; ++mt)
                #pragma unroll
                for (int kc = 0; kc < 4; ++kc) {
                    int k0 = kc * 32 + hi * 8;
                    a2[mt][kc] = lds_frag(&Hw[mt][m][k0], &Hw[mt][m][k0 + 16]);
                }

            int jt0, jt1;
            if      (phase == 0) { jt0 = 0; jt1 = 4; }   // W2x: cols 0..63
            else if (phase == 1) { jt0 = 5; jt1 = 6; }   // W2p: col 80 (padded tile)
            else                 { jt0 = 4; jt1 = 5; }   // W2e: cols 64..79

            for (int jt = jt0; jt < jt1; ++jt) {
                const int n2 = jt * 16 + m;
                v8f c2[MT] = {};
                const bf16_t* wrow2 = sW2t[n2];
                #pragma unroll
                for (int kc = 0; kc < 4; ++kc) {
                    int k0 = kc * 32 + hi * 8;
                    v16bf bF = lds_frag(&wrow2[k0], &wrow2[k0 + 16]);
                    #pragma unroll
                    for (int mt = 0; mt < MT; ++mt)
                        c2[mt] = __builtin_amdgcn_wmma_f32_16x16x32_bf16(
                                    false, a2[mt][kc], false, bF, (short)0, c2[mt], false, false);
                }
                const float bias2 = sB2[n2];

                if (phase == 0) {              // scatter msg_x -> aggregated_x[col]
                    const int f = jt * 16 + m;
                    #pragma unroll
                    for (int mt = 0; mt < MT; ++mt)
                        #pragma unroll
                        for (int r = 0; r < 8; ++r) {
                            int mm = hi * 8 + r;
                            int ee = ebase0 + mt * 16 + mm;
                            if (ee < E_EDGES)
                                atomicAdd(&aggx[(size_t)Colw[mt][mm] * OUT_D + f],
                                          c2[mt][r] + bias2);
                        }
                } else if (phase == 2) {       // edge_update direct store
                    const int f = m;
                    #pragma unroll
                    for (int mt = 0; mt < MT; ++mt)
                        #pragma unroll
                        for (int r = 0; r < 8; ++r) {
                            int mm = hi * 8 + r;
                            int ee = ebase0 + mt * 16 + mm;
                            if (ee < E_EDGES)
                                edgeu[(size_t)ee * BOND_D + f] = c2[mt][r] + bias2;
                        }
                } else {                       // pos: w * rel_pos -> aggregated_pos[col]
                    if (m == 0) {              // lanes 0 & 16 hold the valid column (n2==80)
                        #pragma unroll
                        for (int mt = 0; mt < MT; ++mt)
                            #pragma unroll
                            for (int r = 0; r < 8; ++r) {
                                int mm = hi * 8 + r;
                                int ee = ebase0 + mt * 16 + mm;
                                if (ee < E_EDGES) {
                                    float w = c2[mt][r] + bias2;
                                    int cc = Colw[mt][mm];
                                    atomicAdd(&aggp[(size_t)cc * 3 + 0], w * Relw[mt][mm][0]);
                                    atomicAdd(&aggp[(size_t)cc * 3 + 1], w * Relw[mt][mm][1]);
                                    atomicAdd(&aggp[(size_t)cc * 3 + 2], w * Relw[mt][mm][2]);
                                }
                            }
                    }
                }
            }
        }
    }
}

extern "C" void kernel_launch(void* const* d_in, const int* in_sizes, int n_in,
                              void* d_out, int out_size, void* d_ws, size_t ws_size,
                              hipStream_t stream) {
    const float* x         = (const float*)d_in[0];
    const float* pos       = (const float*)d_in[1];
    const float* edge_attr = (const float*)d_in[2];
    const int*   edge_idx  = (const int*)  d_in[3];
    const float* W1x = (const float*)d_in[4];  const float* b1x = (const float*)d_in[5];
    const float* W2x = (const float*)d_in[6];  const float* b2x = (const float*)d_in[7];
    const float* W1p = (const float*)d_in[8];  const float* b1p = (const float*)d_in[9];
    const float* W2p = (const float*)d_in[10]; const float* b2p = (const float*)d_in[11];
    const float* W1e = (const float*)d_in[12]; const float* b1e = (const float*)d_in[13];
    const float* W2e = (const float*)d_in[14]; const float* b2e = (const float*)d_in[15];

    float* out   = (float*)d_out;
    float* aggx  = out;                                  // [N, 64]
    float* aggp  = out + (size_t)N_NODES * IN_D;         // [N, 3]
    float* edgeu = aggp + (size_t)N_NODES * 3;           // [E, 16]

    // aggregation targets must start from zero every call (d_out is poisoned)
    const int nzero4 = (N_NODES * IN_D + N_NODES * 3) / 4;   // 6.7M divisible by 4
    egnn_zero_kernel<<<2048, BLOCK, 0, stream>>>((float4*)out, nzero4);

    egnn_edge_kernel<<<1024, BLOCK, 0, stream>>>(
        x, pos, edge_attr, edge_idx,
        W1x, b1x, W2x, b2x,
        W1p, b1p, W2p, b2p,
        W1e, b1e, W2e, b2e,
        aggx, aggp, edgeu);
}
